// NSAAttentionExtended_41231686041988
// MI455X (gfx1250) — matrix operand order repaired and checked by reading, not compiled, not run on hardware
//
#include <hip/hip_runtime.h>

typedef __bf16 bf16;
typedef __attribute__((ext_vector_type(16))) __bf16 v16bf;
typedef __attribute__((ext_vector_type(8)))  float  v8f;

#define HIDN  1024
#define SEQ   4096
#define BATCH 4
#define LC    1024      // compressed token count per batch
#define KSEL  512       // top-k selected tokens
#define WIN   256       // window length
#define NWIN  16        // only first 16 windows survive truncation
#define HID3  3072
#define K4    4096
#define ATT_SCALE 0.125f  // 1/sqrt(64)

union FragU { v16bf v; uint4 u[2]; };

// ---------------------------------------------------------------------------
// Generic batched NT GEMM:  out[b,w] = A[b,w] (M x K, ldA) * Bt[b,w]^T + bias
// Bt is stored N x K (ldB). Block tile 128x128, 8 waves, wave tile 32x64.
// OUTB: 0 = f32 row-major out, 1 = bf16 row-major out.
// TRANS: store bf16 transposed: out[col*ldT + row] (per-batch block).
// ACC: f32 out accumulates (out += result).
// ---------------------------------------------------------------------------
template<int OUTB, bool TRANS, bool ACC>
__global__ __launch_bounds__(256)
void gemm_k(const bf16* __restrict__ A,  long long sAb, long long wAoff,
            const bf16* __restrict__ Bt, long long sBb, long long wBoff,
            void* __restrict__ Out,      long long sOb, long long wOoff,
            const float* __restrict__ bias,
            int N, int K, int ldA, int ldB, int ldT, int nwin, float alpha)
{
  const int z = blockIdx.z;
  const int b = z / nwin;
  const int w = z - b * nwin;
  const bf16* Ap = A  + (long long)b * sAb + (long long)w * wAoff;
  const bf16* Bp = Bt + (long long)b * sBb + (long long)w * wBoff;
  const long long obase = (long long)b * sOb + (long long)w * wOoff;

  const int tid  = threadIdx.x;
  const int lane = tid & 31;
  const int wave = tid >> 5;
  const int wm   = wave & 3;       // 4 waves along M
  const int wn   = wave >> 2;      // 2 waves along N
  const int half = lane >> 4;
  const int l16  = lane & 15;
  const int rowBase = blockIdx.y * 128 + wm * 32;
  const int colBase = blockIdx.x * 128 + wn * 64;

  const v8f vzero = {0.f,0.f,0.f,0.f,0.f,0.f,0.f,0.f};
  v8f acc[2][4];
#pragma unroll
  for (int i = 0; i < 2; ++i)
#pragma unroll
    for (int j = 0; j < 4; ++j) acc[i][j] = vzero;

  for (int k0 = 0; k0 < K; k0 += 32) {
    if (k0 + 32 < K) {  // gfx1250 global prefetch of next K-tile lines
      __builtin_prefetch((const void*)(Ap + (long long)(rowBase + l16) * ldA + k0 + 32), 0, 1);
      __builtin_prefetch((const void*)(Bp + (long long)(colBase + l16) * ldB + k0 + 32), 0, 1);
    }
    FragU a[2], bb[4];
#pragma unroll
    for (int i = 0; i < 2; ++i) {
      // A fragment 16x32: lane half 0 -> K k0+0..7 & k0+16..23, half 1 -> +8
      const bf16* p = Ap + (long long)(rowBase + i * 16 + l16) * ldA + k0 + half * 8;
      a[i].u[0] = *(const uint4*)(p);
      a[i].u[1] = *(const uint4*)(p + 16);
    }
#pragma unroll
    for (int j = 0; j < 4; ++j) {
      // B fragment 32x16 (from NxK storage): lane half 0 -> K k0+0..7 & k0+16..23, half 1 -> +8
      const bf16* p = Bp + (long long)(colBase + j * 16 + l16) * ldB + k0 + half * 8;
      bb[j].u[0] = *(const uint4*)(p);
      bb[j].u[1] = *(const uint4*)(p + 16);
    }
#pragma unroll
    for (int i = 0; i < 2; ++i)
#pragma unroll
      for (int j = 0; j < 4; ++j)
        acc[i][j] = __builtin_amdgcn_wmma_f32_16x16x32_bf16(
            false, a[i].v, false, bb[j].v, (short)0, acc[i][j], false, false);
  }

#pragma unroll
  for (int i = 0; i < 2; ++i) {
#pragma unroll
    for (int j = 0; j < 4; ++j) {
      const int col = colBase + j * 16 + l16;
      const float bv = bias ? bias[col] : 0.0f;
#pragma unroll
      for (int r = 0; r < 8; ++r) {
        const int row = rowBase + i * 16 + half * 8 + r;  // C/D layout: lanes16-31 -> M+8
        float v = acc[i][j][r] * alpha + bv;
        if (TRANS) {
          ((bf16*)Out)[obase + (long long)col * ldT + row] = (bf16)v;
        } else if (OUTB == 1) {
          ((bf16*)Out)[obase + (long long)row * N + col] = (bf16)v;
        } else {
          float* o = (float*)Out + obase + (long long)row * N + col;
          float acc_v = v;
          if (ACC) acc_v += *o;
          *o = acc_v;
        }
      }
    }
  }
}

// ---------------------------------------------------------------------------
// Elementwise / helper kernels
// ---------------------------------------------------------------------------
__global__ void cvt_k(const float* __restrict__ src, bf16* __restrict__ dst, int n)
{
  int i = blockIdx.x * blockDim.x + threadIdx.x;
  if (i < n) dst[i] = (bf16)src[i];
}

// WT[n*K + k] = W[k*N + n]   (W is K x N row-major)
__global__ void transpose_k(const float* __restrict__ W, bf16* __restrict__ WT, int K, int N)
{
  int i = blockIdx.x * blockDim.x + threadIdx.x;
  if (i >= K * N) return;
  int n = i / K;
  int k = i - n * K;
  WT[i] = (bf16)W[(long long)k * N + n];
}

__device__ __forceinline__ float blk_sum(float v, float* red)
{
  for (int m = 16; m >= 1; m >>= 1) v += __shfl_xor(v, m, 32);
  const int lane = threadIdx.x & 31, wv = threadIdx.x >> 5;
  if (lane == 0) red[wv] = v;
  __syncthreads();
  float r = 0.f;
  for (int i = 0; i < 8; ++i) r += red[i];
  __syncthreads();
  return r;
}

__device__ __forceinline__ float blk_max(float v, float* red)
{
  for (int m = 16; m >= 1; m >>= 1) v = fmaxf(v, __shfl_xor(v, m, 32));
  const int lane = threadIdx.x & 31, wv = threadIdx.x >> 5;
  if (lane == 0) red[wv] = v;
  __syncthreads();
  float r = red[0];
  for (int i = 1; i < 8; ++i) r = fmaxf(r, red[i]);
  __syncthreads();
  return r;
}

// gates (sigmoid, normalized) and selection scores; one wave per token
__global__ void gates_scores_k(const float* __restrict__ x,
                               const float* __restrict__ Wg, const float* __restrict__ bg,
                               const float* __restrict__ Ws, const float* __restrict__ bs,
                               float* __restrict__ gates, float* __restrict__ scores)
{
  const int tok  = blockIdx.x * (blockDim.x >> 5) + (threadIdx.x >> 5);
  const int lane = threadIdx.x & 31;
  if (tok >= BATCH * SEQ) return;
  const float* xr = x + (long long)tok * HIDN;
  float a0 = 0.f, a1 = 0.f, a2 = 0.f, a3 = 0.f;
  for (int h = lane; h < HIDN; h += 32) {
    float xv = xr[h];
    a0 += xv * Wg[h * 3 + 0];
    a1 += xv * Wg[h * 3 + 1];
    a2 += xv * Wg[h * 3 + 2];
    a3 += xv * Ws[h];
  }
  for (int m = 16; m >= 1; m >>= 1) {
    a0 += __shfl_xor(a0, m, 32);
    a1 += __shfl_xor(a1, m, 32);
    a2 += __shfl_xor(a2, m, 32);
    a3 += __shfl_xor(a3, m, 32);
  }
  if (lane == 0) {
    float g0 = 1.0f / (1.0f + expf(-(a0 + bg[0])));
    float g1 = 1.0f / (1.0f + expf(-(a1 + bg[1])));
    float g2 = 1.0f / (1.0f + expf(-(a2 + bg[2])));
    float inv = 1.0f / (g0 + g1 + g2 + 1e-6f);
    gates[tok * 3 + 0] = g0 * inv;
    gates[tok * 3 + 1] = g1 * inv;
    gates[tok * 3 + 2] = g2 * inv;
    scores[tok] = a3 + bs[0];
  }
}

// exact top-k via rank counting; one block (1024 thr) per batch
__global__ __launch_bounds__(1024) void topk_k(const float* __restrict__ scores,
                                               int* __restrict__ idx)
{
  __shared__ float s[SEQ];
  __shared__ unsigned char f[SEQ];
  const int b = blockIdx.x;
  const int tid = threadIdx.x;
  for (int i = tid; i < SEQ; i += 1024) s[i] = scores[b * SEQ + i];
  __syncthreads();
  for (int i = tid; i < SEQ; i += 1024) {
    float si = s[i];
    int rank = 0;
    for (int j = 0; j < SEQ; ++j) {
      float sj = s[j];
      rank += (sj > si) || (sj == si && j < i);   // jax top_k tie-break: lower index wins
    }
    f[i] = (rank < KSEL) ? 1 : 0;
  }
  __syncthreads();
  for (int i = tid; i < SEQ; i += 1024) {
    if (f[i]) {
      int pos = 0;
      for (int j = 0; j < i; ++j) pos += f[j];
      idx[b * KSEL + pos] = i;                    // naturally index-sorted
    }
  }
}

__global__ void gather_k(const bf16* __restrict__ xb, const int* __restrict__ idx,
                         bf16* __restrict__ selx)
{
  const int r = blockIdx.x;           // b*KSEL + p
  const int b = r / KSEL;
  const int i = idx[r];
  const uint4* src = (const uint4*)(xb + ((long long)b * SEQ + i) * HIDN);
  uint4* dst = (uint4*)(selx + (long long)r * HIDN);
  for (int c = threadIdx.x; c < HIDN / 8; c += blockDim.x) dst[c] = src[c];
}

// row softmax (scale already folded into S) -> bf16 P
__global__ __launch_bounds__(256) void softmax_k(const float* __restrict__ S,
                                                 bf16* __restrict__ P, int cols)
{
  __shared__ float red[8];
  const long long base = (long long)blockIdx.x * cols;
  const int tid = threadIdx.x;
  const int cnt = cols >> 8;          // 1, 2 or 4
  float e[4];
  float mx = -3.4e38f;
  for (int i = 0; i < cnt; ++i) { e[i] = S[base + tid + (i << 8)]; mx = fmaxf(mx, e[i]); }
  mx = blk_max(mx, red);
  float sum = 0.f;
  for (int i = 0; i < cnt; ++i) { e[i] = expf(e[i] - mx); sum += e[i]; }
  sum = blk_sum(sum, red);
  float inv = 1.0f / sum;
  for (int i = 0; i < cnt; ++i) P[base + tid + (i << 8)] = (bf16)(e[i] * inv);
}

// in-place gate scaling of branch outputs (bf16), gate column gidx
__global__ void gate_apply_k(bf16* __restrict__ O, const float* __restrict__ gates,
                             int Lb, int gidx)
{
  int idx = blockIdx.x * blockDim.x + threadIdx.x;
  if (idx >= BATCH * Lb * HIDN) return;
  int row = idx >> 10;                // /HIDN
  int b = row / Lb;
  int t = row - b * Lb;
  float g = gates[((long long)b * SEQ + t) * 3 + gidx];
  O[idx] = (bf16)((float)O[idx] * g);
}

// window output: Aw[b][t] = Ow[b*16 + t/256][t%256] * g2[b][t]
__global__ void build_aw_k(const bf16* __restrict__ Ow, const float* __restrict__ gates,
                           bf16* __restrict__ Aw)
{
  int idx = blockIdx.x * blockDim.x + threadIdx.x;
  if (idx >= BATCH * SEQ * HIDN) return;
  int h = idx & (HIDN - 1);
  int t = (idx >> 10) & (SEQ - 1);
  int b = idx >> 22;
  int w = t >> 8;
  int j = t & (WIN - 1);
  float src = (float)Ow[(((long long)(b * NWIN + w)) * WIN + j) * HIDN + h];
  float g = gates[((long long)b * SEQ + t) * 3 + 2];
  Aw[idx] = (bf16)(src * g);
}

// out = LN(0.5*lin + 0.5*x), eps 1e-6, no affine
__global__ __launch_bounds__(256) void final_ln_k(const float* __restrict__ lin,
                                                  const float* __restrict__ x,
                                                  float* __restrict__ out)
{
  __shared__ float red[8];
  const long long base = (long long)blockIdx.x * HIDN;
  const int tid = threadIdx.x;
  float v[4];
  float s = 0.f, s2 = 0.f;
  for (int i = 0; i < 4; ++i) {
    float t = 0.5f * lin[base + tid + (i << 8)] + 0.5f * x[base + tid + (i << 8)];
    v[i] = t; s += t; s2 += t * t;
  }
  s  = blk_sum(s,  red);
  s2 = blk_sum(s2, red);
  float mean = s * (1.0f / HIDN);
  float var  = s2 * (1.0f / HIDN) - mean * mean;
  float inv  = rsqrtf(var + 1e-6f);
  for (int i = 0; i < 4; ++i) out[base + tid + (i << 8)] = (v[i] - mean) * inv;
}

// ---------------------------------------------------------------------------
extern "C" void kernel_launch(void* const* d_in, const int* in_sizes, int n_in,
                              void* d_out, int out_size, void* d_ws, size_t ws_size,
                              hipStream_t stream)
{
  (void)in_sizes; (void)n_in; (void)out_size; (void)ws_size;
  const float* x  = (const float*)d_in[0];
  const float* Wq = (const float*)d_in[1];
  const float* bq = (const float*)d_in[2];
  const float* Wk = (const float*)d_in[3];
  const float* bk = (const float*)d_in[4];
  const float* Wv = (const float*)d_in[5];
  const float* bv = (const float*)d_in[6];
  const float* Wo = (const float*)d_in[7];
  const float* bo = (const float*)d_in[8];
  const float* Wg = (const float*)d_in[9];
  const float* bg = (const float*)d_in[10];
  const float* Wc = (const float*)d_in[11];
  const float* bc = (const float*)d_in[12];
  const float* Ws = (const float*)d_in[13];
  const float* bs = (const float*)d_in[14];
  float* out = (float*)d_out;

  char* wsb = (char*)d_ws;
  size_t off = 0;
  auto alloc = [&](size_t bytes) -> void* {
    void* p = wsb + off;
    off += (bytes + 255) & ~(size_t)255;
    return p;
  };

  const long long LH = (long long)BATCH * SEQ * HIDN;   // 16.7M elements

  bf16* xb   = (bf16*)alloc((size_t)LH * 2);
  bf16* WqT  = (bf16*)alloc((size_t)HIDN * HIDN * 2);
  bf16* WkT  = (bf16*)alloc((size_t)HIDN * HIDN * 2);
  bf16* WvT  = (bf16*)alloc((size_t)HIDN * HIDN * 2);
  bf16* WcT  = (bf16*)alloc((size_t)HIDN * K4 * 2);
  bf16* WoT  = (bf16*)alloc((size_t)HIDN * HID3 * 2);
  float* gates  = (float*)alloc((size_t)BATCH * SEQ * 3 * 4);
  float* scores = (float*)alloc((size_t)BATCH * SEQ * 4);
  int*   idx    = (int*)alloc((size_t)BATCH * KSEL * 4);
  bf16* selx = (bf16*)alloc((size_t)BATCH * KSEL * HIDN * 2);
  bf16* cmp  = (bf16*)alloc((size_t)BATCH * LC * HIDN * 2);
  bf16* qc   = (bf16*)alloc((size_t)BATCH * LC * HIDN * 2);
  bf16* kc   = (bf16*)alloc((size_t)BATCH * LC * HIDN * 2);
  bf16* vcT  = (bf16*)alloc((size_t)BATCH * LC * HIDN * 2);
  float* Sc  = (float*)alloc((size_t)BATCH * LC * LC * 4);
  bf16* Pc   = (bf16*)alloc((size_t)BATCH * LC * LC * 2);
  bf16* Oc   = (bf16*)alloc((size_t)BATCH * LC * HIDN * 2);
  bf16* qs   = (bf16*)alloc((size_t)BATCH * KSEL * HIDN * 2);
  bf16* ks   = (bf16*)alloc((size_t)BATCH * KSEL * HIDN * 2);
  bf16* vsT  = (bf16*)alloc((size_t)BATCH * KSEL * HIDN * 2);
  float* Ss  = (float*)alloc((size_t)BATCH * KSEL * KSEL * 4);
  bf16* Ps   = (bf16*)alloc((size_t)BATCH * KSEL * KSEL * 2);
  bf16* Os   = (bf16*)alloc((size_t)BATCH * KSEL * HIDN * 2);
  bf16* qw   = (bf16*)alloc((size_t)LH * 2);
  bf16* kw   = (bf16*)alloc((size_t)LH * 2);
  bf16* vwT  = (bf16*)alloc((size_t)LH * 2);
  float* Sw  = (float*)alloc((size_t)BATCH * NWIN * WIN * WIN * 4);
  bf16* Pw   = (bf16*)alloc((size_t)BATCH * NWIN * WIN * WIN * 2);
  bf16* Ow   = (bf16*)alloc((size_t)BATCH * NWIN * WIN * HIDN * 2);
  bf16* Aw   = (bf16*)alloc((size_t)LH * 2);
  float* lin = (float*)alloc((size_t)LH * 4);

  const int T = 256;

  // ---- stage 0: conversions & weight transposes -------------------------
  cvt_k<<<(int)((LH + T - 1) / T), T, 0, stream>>>(x, xb, (int)LH);
  transpose_k<<<(HIDN * HIDN + T - 1) / T, T, 0, stream>>>(Wq, WqT, HIDN, HIDN);
  transpose_k<<<(HIDN * HIDN + T - 1) / T, T, 0, stream>>>(Wk, WkT, HIDN, HIDN);
  transpose_k<<<(HIDN * HIDN + T - 1) / T, T, 0, stream>>>(Wv, WvT, HIDN, HIDN);
  transpose_k<<<(K4 * HIDN + T - 1) / T, T, 0, stream>>>(Wc, WcT, K4, HIDN);
  transpose_k<<<(HID3 * HIDN + T - 1) / T, T, 0, stream>>>(Wo, WoT, HID3, HIDN);

  // ---- gates / scores / top-k / gather ----------------------------------
  gates_scores_k<<<(BATCH * SEQ + 7) / 8, 256, 0, stream>>>(x, Wg, bg, Ws, bs, gates, scores);
  topk_k<<<BATCH, 1024, 0, stream>>>(scores, idx);
  gather_k<<<BATCH * KSEL, 128, 0, stream>>>(xb, idx, selx);

  // ---- compression GEMM: blocks(B,1024,4096) = xb reinterpreted ---------
  gemm_k<1, false, false><<<dim3(HIDN / 128, LC / 128, BATCH), 256, 0, stream>>>(
      xb, (long long)SEQ * HIDN, 0, WcT, 0, 0, cmp, (long long)LC * HIDN, 0, bc,
      HIDN, K4, K4, K4, 0, 1, 1.0f);

  // ---- QKV projections (V stored transposed: dims x tokens) -------------
  // compressed branch
  gemm_k<1, false, false><<<dim3(8, LC / 128, BATCH), 256, 0, stream>>>(
      cmp, (long long)LC * HIDN, 0, WqT, 0, 0, qc, (long long)LC * HIDN, 0, bq,
      HIDN, HIDN, HIDN, HIDN, 0, 1, 1.0f);
  gemm_k<1, false, false><<<dim3(8, LC / 128, BATCH), 256, 0, stream>>>(
      cmp, (long long)LC * HIDN, 0, WkT, 0, 0, kc, (long long)LC * HIDN, 0, bk,
      HIDN, HIDN, HIDN, HIDN, 0, 1, 1.0f);
  gemm_k<1, true, false><<<dim3(8, LC / 128, BATCH), 256, 0, stream>>>(
      cmp, (long long)LC * HIDN, 0, WvT, 0, 0, vcT, (long long)LC * HIDN, 0, bv,
      HIDN, HIDN, HIDN, HIDN, LC, 1, 1.0f);
  // selected branch
  gemm_k<1, false, false><<<dim3(8, KSEL / 128, BATCH), 256, 0, stream>>>(
      selx, (long long)KSEL * HIDN, 0, WqT, 0, 0, qs, (long long)KSEL * HIDN, 0, bq,
      HIDN, HIDN, HIDN, HIDN, 0, 1, 1.0f);
  gemm_k<1, false, false><<<dim3(8, KSEL / 128, BATCH), 256, 0, stream>>>(
      selx, (long long)KSEL * HIDN, 0, WkT, 0, 0, ks, (long long)KSEL * HIDN, 0, bk,
      HIDN, HIDN, HIDN, HIDN, 0, 1, 1.0f);
  gemm_k<1, true, false><<<dim3(8, KSEL / 128, BATCH), 256, 0, stream>>>(
      selx, (long long)KSEL * HIDN, 0, WvT, 0, 0, vsT, (long long)KSEL * HIDN, 0, bv,
      HIDN, HIDN, HIDN, HIDN, KSEL, 1, 1.0f);
  // window branch (per original token, shared by all windows)
  gemm_k<1, false, false><<<dim3(8, SEQ / 128, BATCH), 256, 0, stream>>>(
      xb, (long long)SEQ * HIDN, 0, WqT, 0, 0, qw, (long long)SEQ * HIDN, 0, bq,
      HIDN, HIDN, HIDN, HIDN, 0, 1, 1.0f);
  gemm_k<1, false, false><<<dim3(8, SEQ / 128, BATCH), 256, 0, stream>>>(
      xb, (long long)SEQ * HIDN, 0, WkT, 0, 0, kw, (long long)SEQ * HIDN, 0, bk,
      HIDN, HIDN, HIDN, HIDN, 0, 1, 1.0f);
  gemm_k<1, true, false><<<dim3(8, SEQ / 128, BATCH), 256, 0, stream>>>(
      xb, (long long)SEQ * HIDN, 0, WvT, 0, 0, vwT, (long long)SEQ * HIDN, 0, bv,
      HIDN, HIDN, HIDN, HIDN, SEQ, 1, 1.0f);

  // ---- compressed attention --------------------------------------------
  gemm_k<0, false, false><<<dim3(LC / 128, LC / 128, BATCH), 256, 0, stream>>>(
      qc, (long long)LC * HIDN, 0, kc, (long long)LC * HIDN, 0,
      Sc, (long long)LC * LC, 0, nullptr, LC, HIDN, HIDN, HIDN, 0, 1, ATT_SCALE);
  softmax_k<<<BATCH * LC, 256, 0, stream>>>(Sc, Pc, LC);
  gemm_k<1, false, false><<<dim3(HIDN / 128, LC / 128, BATCH), 256, 0, stream>>>(
      Pc, (long long)LC * LC, 0, vcT, (long long)LC * HIDN, 0,
      Oc, (long long)LC * HIDN, 0, nullptr, HIDN, LC, LC, LC, 0, 1, 1.0f);
  gate_apply_k<<<(BATCH * LC * HIDN + T - 1) / T, T, 0, stream>>>(Oc, gates, LC, 0);

  // ---- selected attention -----------------------------------------------
  gemm_k<0, false, false><<<dim3(KSEL / 128, KSEL / 128, BATCH), 256, 0, stream>>>(
      qs, (long long)KSEL * HIDN, 0, ks, (long long)KSEL * HIDN, 0,
      Ss, (long long)KSEL * KSEL, 0, nullptr, KSEL, HIDN, HIDN, HIDN, 0, 1, ATT_SCALE);
  softmax_k<<<BATCH * KSEL, 256, 0, stream>>>(Ss, Ps, KSEL);
  gemm_k<1, false, false><<<dim3(HIDN / 128, KSEL / 128, BATCH), 256, 0, stream>>>(
      Ps, (long long)KSEL * KSEL, 0, vsT, (long long)KSEL * HIDN, 0,
      Os, (long long)KSEL * HIDN, 0, nullptr, HIDN, KSEL, KSEL, KSEL, 0, 1, 1.0f);
  gate_apply_k<<<(BATCH * KSEL * HIDN + T - 1) / T, T, 0, stream>>>(Os, gates, KSEL, 1);

  // ---- window attention (16 windows x 256 tokens, start step 128) -------
  gemm_k<0, false, false><<<dim3(WIN / 128, WIN / 128, BATCH * NWIN), 256, 0, stream>>>(
      qw, (long long)SEQ * HIDN, (long long)128 * HIDN,
      kw, (long long)SEQ * HIDN, (long long)128 * HIDN,
      Sw, (long long)NWIN * WIN * WIN, (long long)WIN * WIN,
      nullptr, WIN, HIDN, HIDN, HIDN, 0, NWIN, ATT_SCALE);
  softmax_k<<<BATCH * NWIN * WIN, 256, 0, stream>>>(Sw, Pw, WIN);
  gemm_k<1, false, false><<<dim3(HIDN / 128, WIN / 128, BATCH * NWIN), 256, 0, stream>>>(
      Pw, (long long)NWIN * WIN * WIN, (long long)WIN * WIN,
      vwT, (long long)SEQ * HIDN, 128,
      Ow, (long long)NWIN * WIN * HIDN, (long long)WIN * HIDN,
      nullptr, HIDN, WIN, WIN, SEQ, 0, NWIN, 1.0f);
  build_aw_k<<<(int)((LH + T - 1) / T), T, 0, stream>>>(Ow, gates, Aw);

  // ---- output projection: split 3H concat into 3 accumulating GEMMs -----
  // lin = Aw @ Wo[2048:3072,:] + bo  (covers all 4096 rows per batch)
  gemm_k<0, false, false><<<dim3(HIDN / 128, SEQ / 128, BATCH), 256, 0, stream>>>(
      Aw, (long long)SEQ * HIDN, 0, WoT + 2048, 0, 0,
      lin, (long long)SEQ * HIDN, 0, bo, HIDN, HIDN, HIDN, HID3, 0, 1, 1.0f);
  // lin[:1024] += Oc @ Wo[0:1024,:]
  gemm_k<0, false, true><<<dim3(HIDN / 128, LC / 128, BATCH), 256, 0, stream>>>(
      Oc, (long long)LC * HIDN, 0, WoT, 0, 0,
      lin, (long long)SEQ * HIDN, 0, nullptr, HIDN, HIDN, HIDN, HID3, 0, 1, 1.0f);
  // lin[:512] += Os @ Wo[1024:2048,:]
  gemm_k<0, false, true><<<dim3(HIDN / 128, KSEL / 128, BATCH), 256, 0, stream>>>(
      Os, (long long)KSEL * HIDN, 0, WoT + 1024, 0, 0,
      lin, (long long)SEQ * HIDN, 0, nullptr, HIDN, HIDN, HIDN, HID3, 0, 1, 1.0f);

  // ---- residual + LayerNorm ---------------------------------------------
  final_ln_k<<<BATCH * SEQ, 256, 0, stream>>>(lin, x, out);
}
